// Mamba2Simple_86414741995728
// MI455X (gfx1250) — matrix operand order repaired and checked by reading, not compiled, not run on hardware
//
#include <hip/hip_runtime.h>
#include <hip/hip_bf16.h>
#include <cstdint>

#define D_MODEL   2048
#define D_STATE   64
#define HEADDIM   128
#define CHUNK     256
#define D_INNER   4096
#define NHEADS    32
#define CONV_DIM  4224
#define D_IN_PROJ 8352
#define BATCH     2
#define SEQLEN    4096
#define NCHUNK    16
#define ROWS      (BATCH*SEQLEN)   /* 8192 */
#define LN_EPS    1e-5f

typedef __attribute__((ext_vector_type(16))) __bf16 bf16x16;
typedef __attribute__((ext_vector_type(8)))  float  f32x8;

__device__ __forceinline__ unsigned short f2bf(float f) {
  union { float f; unsigned u; } c; c.f = f;
  unsigned r = c.u + 0x7FFFu + ((c.u >> 16) & 1u);
  return (unsigned short)(r >> 16);
}
__device__ __forceinline__ float bf2f(unsigned short h) {
  union { unsigned u; float f; } c; c.u = ((unsigned)h) << 16; return c.f;
}
__device__ __forceinline__ float siluf_(float x) { return x / (1.0f + __expf(-x)); }

__device__ __forceinline__ f32x8 wmma_bf16(bf16x16 a, bf16x16 b, f32x8 c) {
  return __builtin_amdgcn_wmma_f32_16x16x32_bf16(false, a, false, b, (short)0, c,
                                                 false, false);
}

union FragB { uint4 q[2]; bf16x16 v; };
union FragS { unsigned short us[16]; bf16x16 v; };

#define LDT 40  /* halves per LDS row: 80B, 16B-aligned */

// ---- CDNA5 async global->LDS copy (ASYNCcnt path), with sync fallback ------
#if defined(__has_builtin)
#if __has_builtin(__builtin_amdgcn_global_load_async_to_lds_b128)
#define HAVE_ASYNC_LDS 1
#endif
#endif

typedef int v4i __attribute__((vector_size(16)));
typedef __attribute__((address_space(1))) v4i* gas_b128;
typedef __attribute__((address_space(3))) v4i* las_b128;

__device__ __forceinline__ void cp_lds_b128(unsigned short* lds_dst,
                                            const unsigned short* gsrc) {
#ifdef HAVE_ASYNC_LDS
  __builtin_amdgcn_global_load_async_to_lds_b128(
      (gas_b128)(gsrc), (las_b128)(lds_dst), 0, 0);
#else
  *(uint4*)lds_dst = *(const uint4*)gsrc;
#endif
}
__device__ __forceinline__ void wait_async() {
#ifdef HAVE_ASYNC_LDS
#if __has_builtin(__builtin_amdgcn_s_wait_asynccnt)
  __builtin_amdgcn_s_wait_asynccnt(0);
#else
  asm volatile("s_wait_asynccnt 0x0" ::: "memory");
#endif
#endif
}

// ---------------------------------------------------------------------------
// fp32 -> bf16 bulk conversion (n4 = count/4)
// ---------------------------------------------------------------------------
__global__ __launch_bounds__(256) void k_cvt(const float* __restrict__ src,
                                             unsigned short* __restrict__ dst,
                                             size_t n4)
{
  size_t i = (size_t)blockIdx.x * 256 + threadIdx.x;
  if (i >= n4) return;
  float4 v = ((const float4*)src)[i];
  unsigned lo = (unsigned)f2bf(v.x) | ((unsigned)f2bf(v.y) << 16);
  unsigned hi = (unsigned)f2bf(v.z) | ((unsigned)f2bf(v.w) << 16);
  ((uint2*)dst)[i] = make_uint2(lo, hi);
}

// ---------------------------------------------------------------------------
// bf16 GEMM: out[M,N] = A[M,K] * W[N,K]^T + bias[N]   (M%128==0, K%32==0)
// block tile 128x128, 8 waves (2x4), wave tile 64x32 (4x2 frags), K-step 32.
// Double-buffered async global->LDS pipeline: prefetch k+1, compute k, wait.
// ---------------------------------------------------------------------------
__global__ __launch_bounds__(256) void k_gemm_bf(
    const unsigned short* __restrict__ A, int lda,
    const unsigned short* __restrict__ W, int ldw,
    const float* __restrict__ bias,
    float* __restrict__ out, int ldo,
    int M, int N, int K)
{
  __shared__ alignas(16) unsigned short As[2][128 * LDT];
  __shared__ alignas(16) unsigned short Ws[2][128 * LDT];
  const int tid = threadIdx.x, lane = tid & 31, wave = tid >> 5;
  const int wm0 = (wave >> 2) * 64, wn0 = (wave & 3) * 32;
  const int m0 = blockIdx.y * 128, n0 = blockIdx.x * 128;
  const int abase = (lane < 16) ? 0 : 8;
  const int bbase = (lane < 16) ? 0 : 8;

  f32x8 acc[4][2];
  for (int i = 0; i < 4; ++i)
    for (int j = 0; j < 2; ++j)
      for (int r = 0; r < 8; ++r) acc[i][j][r] = 0.0f;

  // per-thread copy coordinates (512 b128 chunks per 128x32 tile, 2/thread)
  auto fill = [&](int buf, int k0) {
    for (int i = 0; i < 2; ++i) {
      int ch = tid + i * 256;
      int rr = ch >> 2, k8 = (ch & 3) * 8;
      cp_lds_b128(&As[buf][rr * LDT + k8], &A[(size_t)(m0 + rr) * lda + k0 + k8]);
      int gn = n0 + rr; if (gn > N - 1) gn = N - 1;  // clamp ragged N edge
      cp_lds_b128(&Ws[buf][rr * LDT + k8], &W[(size_t)gn * ldw + k0 + k8]);
    }
  };

  fill(0, 0);
  wait_async();
  __syncthreads();

  int cur = 0;
  for (int k0 = 0; k0 < K; k0 += 32) {
    if (k0 + 32 < K) fill(cur ^ 1, k0 + 32);  // prefetch next tile (async)

    FragB bf[2];
    for (int j = 0; j < 2; ++j) {
      const unsigned short* p = &Ws[cur][(wn0 + j * 16 + (lane & 15)) * LDT + bbase];
      bf[j].q[0] = *(const uint4*)(p);
      bf[j].q[1] = *(const uint4*)(p + 16);
    }
    for (int i = 0; i < 4; ++i) {
      FragB af;
      const unsigned short* p = &As[cur][(wm0 + i * 16 + (lane & 15)) * LDT + abase];
      af.q[0] = *(const uint4*)(p);
      af.q[1] = *(const uint4*)(p + 16);
      for (int j = 0; j < 2; ++j)
        acc[i][j] = wmma_bf16(af.v, bf[j].v, acc[i][j]);
    }

    wait_async();      // own async copies into buf^1 have landed
    __syncthreads();   // everyone done reading cur / writing nxt
    cur ^= 1;
  }

  const int moff = (lane < 16) ? 0 : 8;
  for (int i = 0; i < 4; ++i)
    for (int j = 0; j < 2; ++j) {
      int ncol = n0 + wn0 + j * 16 + (lane & 15);
      if (ncol < N) {
        float bv = bias[ncol];
        int mbase = m0 + wm0 + i * 16 + moff;
        for (int r = 0; r < 8; ++r)
          out[(size_t)(mbase + r) * ldo + ncol] = acc[i][j][r] + bv;
      }
    }
}

// ---------------------------------------------------------------------------
// depthwise causal conv1d (k=4) + bias + silu
// ---------------------------------------------------------------------------
__global__ __launch_bounds__(256) void k_conv(
    const float* __restrict__ zx, const float* __restrict__ cw,
    const float* __restrict__ cb, float* __restrict__ xbc)
{
  size_t i = (size_t)blockIdx.x * 256 + threadIdx.x;
  if (i >= (size_t)ROWS * CONV_DIM) return;
  int c = (int)(i % CONV_DIM);
  size_t row = i / CONV_DIM;
  int l = (int)(row % SEQLEN);
  float acc = cb[c];
  for (int j = 0; j < 4; ++j) {
    int ls = l - 3 + j;
    if (ls >= 0)
      acc += cw[c * 4 + j] * zx[(row - 3 + j) * (size_t)D_IN_PROJ + D_INNER + c];
  }
  xbc[i] = siluf_(acc);
}

__global__ __launch_bounds__(256) void k_dt(
    const float* __restrict__ zx, const float* __restrict__ dt_bias,
    float* __restrict__ dt)
{
  int i = blockIdx.x * 256 + threadIdx.x;
  if (i >= ROWS * NHEADS) return;
  int h = i & (NHEADS - 1), row = i >> 5;
  float x = zx[(size_t)row * D_IN_PROJ + D_INNER + CONV_DIM + h] + dt_bias[h];
  dt[i] = (x > 20.0f) ? x : log1pf(__expf(x));
}

// per-(b,h,chunk) inclusive scan of dA = -exp(A_log)*dt
__global__ __launch_bounds__(256) void k_acs(
    const float* __restrict__ dt, const float* __restrict__ A_log,
    float* __restrict__ acs, float* __restrict__ alast)
{
  int blk = blockIdx.x;  // (b*NHEADS + h)*NCHUNK + c
  int c = blk % NCHUNK, h = (blk / NCHUNK) % NHEADS, b = blk / (NCHUNK * NHEADS);
  int t = threadIdx.x;
  __shared__ float s[2][CHUNK];
  float v = -__expf(A_log[h]) * dt[(size_t)(b * SEQLEN + c * CHUNK + t) * NHEADS + h];
  int src = 0;
  s[0][t] = v; __syncthreads();
  for (int off = 1; off < CHUNK; off <<= 1) {
    float x = s[src][t];
    if (t >= off) x += s[src][t - off];
    s[src ^ 1][t] = x; __syncthreads();
    src ^= 1;
  }
  float cs = s[src][t];
  acs[(size_t)blk * CHUNK + t] = cs;
  if (t == CHUNK - 1) alast[blk] = cs;
}

// Xdt[b,l,h,p] = bf16(x * dt)
__global__ __launch_bounds__(256) void k_xdt(
    const float* __restrict__ xbc, const float* __restrict__ dt,
    unsigned short* __restrict__ xdt)
{
  size_t i = (size_t)blockIdx.x * 256 + threadIdx.x;
  if (i >= (size_t)ROWS * D_INNER) return;
  int p = (int)(i & (HEADDIM - 1));
  int h = (int)((i >> 7) & (NHEADS - 1));
  size_t row = i >> 12;
  float x = xbc[row * CONV_DIM + h * HEADDIM + p];
  xdt[i] = f2bf(x * dt[row * NHEADS + h]);
}

// ---------------------------------------------------------------------------
// G[b,c,l,s] = sum_n C[l,n]*B[s,n]  (head-independent; NGROUPS==1)
// ---------------------------------------------------------------------------
#define GLD 80
__global__ __launch_bounds__(256) void k_G(
    const float* __restrict__ xbc, float* __restrict__ G)
{
  int bc = blockIdx.x, sq = blockIdx.y;
  const float* base = xbc + (size_t)((bc / NCHUNK) * SEQLEN + (bc % NCHUNK) * CHUNK) * CONV_DIM;
  __shared__ alignas(16) unsigned short Ct[CHUNK * GLD];  // [l][n]
  __shared__ alignas(16) unsigned short Bt[64 * GLD];     // [s][n]
  int tid = threadIdx.x, lane = tid & 31, wave = tid >> 5;
  int s0 = sq * 64;
  for (int i = 0; i < 16; ++i) {          // 4096 float4 chunks
    int ch = tid + i * 256, l = ch >> 4, n4 = (ch & 15) * 4;
    float4 v = *(const float4*)(base + (size_t)l * CONV_DIM + D_INNER + D_STATE + n4);
    unsigned lo = (unsigned)f2bf(v.x) | ((unsigned)f2bf(v.y) << 16);
    unsigned hi = (unsigned)f2bf(v.z) | ((unsigned)f2bf(v.w) << 16);
    *(uint2*)&Ct[l * GLD + n4] = make_uint2(lo, hi);
  }
  for (int i = 0; i < 4; ++i) {           // 1024 float4 chunks
    int ch = tid + i * 256, s = ch >> 4, n4 = (ch & 15) * 4;
    float4 v = *(const float4*)(base + (size_t)(s0 + s) * CONV_DIM + D_INNER + n4);
    unsigned lo = (unsigned)f2bf(v.x) | ((unsigned)f2bf(v.y) << 16);
    unsigned hi = (unsigned)f2bf(v.z) | ((unsigned)f2bf(v.w) << 16);
    *(uint2*)&Bt[s * GLD + n4] = make_uint2(lo, hi);
  }
  __syncthreads();

  const int abase = (lane < 16) ? 0 : 8, bbase = (lane < 16) ? 0 : 8;
  f32x8 acc[2][4];
  for (int i = 0; i < 2; ++i)
    for (int j = 0; j < 4; ++j)
      for (int r = 0; r < 8; ++r) acc[i][j][r] = 0.0f;

  for (int kk = 0; kk < D_STATE; kk += 32) {
    FragB bf[4];
    for (int j = 0; j < 4; ++j) {
      const unsigned short* p = &Bt[(j * 16 + (lane & 15)) * GLD + kk + bbase];
      bf[j].q[0] = *(const uint4*)(p);
      bf[j].q[1] = *(const uint4*)(p + 16);
    }
    for (int i = 0; i < 2; ++i) {
      FragB af;
      const unsigned short* p = &Ct[(wave * 32 + i * 16 + (lane & 15)) * GLD + kk + abase];
      af.q[0] = *(const uint4*)(p);
      af.q[1] = *(const uint4*)(p + 16);
      for (int j = 0; j < 4; ++j)
        acc[i][j] = wmma_bf16(af.v, bf[j].v, acc[i][j]);
    }
  }
  float* g = G + (size_t)bc * CHUNK * CHUNK;
  const int moff = (lane < 16) ? 0 : 8;
  for (int i = 0; i < 2; ++i)
    for (int j = 0; j < 4; ++j) {
      int lbase = wave * 32 + i * 16 + moff;
      int scol = s0 + j * 16 + (lane & 15);
      for (int r = 0; r < 8; ++r)
        g[(size_t)(lbase + r) * CHUNK + scol] = acc[i][j][r];
    }
}

// ---------------------------------------------------------------------------
// chunk states: cs[b,c,h,p,n] = sum_l Xdt[l,p]*exp(Alast-acs[l])*B[l,n]
// ---------------------------------------------------------------------------
__global__ __launch_bounds__(256) void k_cs(
    const unsigned short* __restrict__ xdt, const float* __restrict__ xbc,
    const float* __restrict__ acs, const float* __restrict__ alast,
    float* __restrict__ cs)
{
  int blk = blockIdx.x;  // (b*NCHUNK + c)*NHEADS + h
  int h = blk % NHEADS, c = (blk / NHEADS) % NCHUNK, b = blk / (NHEADS * NCHUNK);
  int tid = threadIdx.x, lane = tid & 31, wave = tid >> 5;
  __shared__ alignas(16) unsigned short At[HEADDIM * LDT];  // [p][l]
  __shared__ alignas(16) unsigned short Bl[D_STATE * LDT];  // [n][l]
  const float* acsp = acs + (size_t)((b * NHEADS + h) * NCHUNK + c) * CHUNK;
  float al = alast[(b * NHEADS + h) * NCHUNK + c];
  const unsigned short* xrow = xdt + ((size_t)(b * SEQLEN + c * CHUNK) * NHEADS + h) * HEADDIM;
  const float* brow = xbc + (size_t)(b * SEQLEN + c * CHUNK) * CONV_DIM + D_INNER;
  const int abase = (lane < 16) ? 0 : 8, bbase = (lane < 16) ? 0 : 8;

  f32x8 acc[4];
  for (int j = 0; j < 4; ++j)
    for (int r = 0; r < 8; ++r) acc[j][r] = 0.0f;

  for (int lt = 0; lt < CHUNK; lt += 32) {
    // At: wide b128 reads along p, transpose-scatter into [p][l], scaled
    for (int i = 0; i < 2; ++i) {
      int ch = tid + i * 256;
      int ll = ch & 31, p8 = (ch >> 5) * 8;
      float dec = __expf(al - acsp[lt + ll]);
      uint4 q = *(const uint4*)(xrow + (size_t)(lt + ll) * D_INNER + p8);
      unsigned short hs[8] = {
        (unsigned short)(q.x & 0xffff), (unsigned short)(q.x >> 16),
        (unsigned short)(q.y & 0xffff), (unsigned short)(q.y >> 16),
        (unsigned short)(q.z & 0xffff), (unsigned short)(q.z >> 16),
        (unsigned short)(q.w & 0xffff), (unsigned short)(q.w >> 16)};
      for (int j = 0; j < 8; ++j)
        At[(p8 + j) * LDT + ll] = f2bf(bf2f(hs[j]) * dec);
    }
    // Bl: float4 reads along n, transpose-scatter into [n][l]
    for (int i = 0; i < 2; ++i) {
      int ch = tid + i * 256;
      int ll = ch & 31, n4 = (ch >> 5) * 4;
      float4 v = *(const float4*)(brow + (size_t)(lt + ll) * CONV_DIM + n4);
      Bl[(n4 + 0) * LDT + ll] = f2bf(v.x);
      Bl[(n4 + 1) * LDT + ll] = f2bf(v.y);
      Bl[(n4 + 2) * LDT + ll] = f2bf(v.z);
      Bl[(n4 + 3) * LDT + ll] = f2bf(v.w);
    }
    __syncthreads();
    FragB af;
    const unsigned short* pa = &At[(wave * 16 + (lane & 15)) * LDT + abase];
    af.q[0] = *(const uint4*)(pa);
    af.q[1] = *(const uint4*)(pa + 16);
    for (int j = 0; j < 4; ++j) {
      FragB bq;
      const unsigned short* pb = &Bl[(j * 16 + (lane & 15)) * LDT + bbase];
      bq.q[0] = *(const uint4*)(pb);
      bq.q[1] = *(const uint4*)(pb + 16);
      acc[j] = wmma_bf16(af.v, bq.v, acc[j]);
    }
    __syncthreads();
  }
  float* csp = cs + (size_t)blk * HEADDIM * D_STATE;
  const int moff = (lane < 16) ? 0 : 8;
  for (int j = 0; j < 4; ++j) {
    int pbase = wave * 16 + moff;
    int ncol = j * 16 + (lane & 15);
    for (int r = 0; r < 8; ++r)
      csp[(size_t)(pbase + r) * D_STATE + ncol] = acc[j][r];
  }
}

// inter-chunk recurrence
__global__ __launch_bounds__(256) void k_recur(
    const float* __restrict__ cs, const float* __restrict__ alast,
    float* __restrict__ R)
{
  int i = blockIdx.x * 256 + threadIdx.x;
  if (i >= BATCH * NHEADS * HEADDIM * D_STATE) return;
  int n = i & 63, p = (i >> 6) & 127, h = (i >> 13) & 31, b = i >> 18;
  float Rv = 0.0f;
  for (int c = 0; c < NCHUNK; ++c) {
    size_t idx = (((size_t)(b * NCHUNK + c) * NHEADS + h) * HEADDIM + p) * D_STATE + n;
    R[idx] = Rv;
    Rv = __expf(alast[(b * NHEADS + h) * NCHUNK + c]) * Rv + cs[idx];
  }
}

// ---------------------------------------------------------------------------
// Y = (G .* L) @ Xdt  +  exp(acs[l]) * C @ R^T     per (b,c,h)
// ---------------------------------------------------------------------------
__global__ __launch_bounds__(256) void k_y(
    const unsigned short* __restrict__ xdt, const float* __restrict__ xbc,
    const float* __restrict__ G, const float* __restrict__ R,
    const float* __restrict__ acs, float* __restrict__ y)
{
  int blk = blockIdx.x;
  int h = blk % NHEADS, c = (blk / NHEADS) % NCHUNK, b = blk / (NHEADS * NCHUNK);
  int tid = threadIdx.x, lane = tid & 31, wave = tid >> 5;
  __shared__ float sacs[CHUNK];
  __shared__ alignas(16) unsigned short Xs[HEADDIM * LDT];  // [p][s]
  __shared__ alignas(16) unsigned short Ms[8][32 * LDT];    // per-wave masked tile
  sacs[tid] = acs[(size_t)((b * NHEADS + h) * NCHUNK + c) * CHUNK + tid];
  __syncthreads();

  const int l0 = wave * 32;
  const int abase = (lane < 16) ? 0 : 8, bbase = (lane < 16) ? 0 : 8;
  f32x8 acc[2][8];
  for (int i = 0; i < 2; ++i)
    for (int j = 0; j < 8; ++j)
      for (int r = 0; r < 8; ++r) acc[i][j][r] = 0.0f;

  // ---- Y_off: A = C scaled by exp(acs[l]); B = R^T (float4 gathers)
  const float* Crow = xbc + (size_t)(b * SEQLEN + c * CHUNK) * CONV_DIM + D_INNER + D_STATE;
  const float* Rp = R + ((size_t)(b * NCHUNK + c) * NHEADS + h) * HEADDIM * D_STATE;
  for (int kk = 0; kk < D_STATE; kk += 32) {
    FragS af[2];
    for (int i = 0; i < 2; ++i) {
      int l = l0 + i * 16 + (lane & 15);
      float sc = __expf(sacs[l]);
      const float* cp = Crow + (size_t)l * CONV_DIM + kk + abase;
      float4 a0 = *(const float4*)(cp), a1 = *(const float4*)(cp + 4);
      float4 a2 = *(const float4*)(cp + 16), a3 = *(const float4*)(cp + 20);
      af[i].us[0] = f2bf(a0.x * sc);  af[i].us[1] = f2bf(a0.y * sc);
      af[i].us[2] = f2bf(a0.z * sc);  af[i].us[3] = f2bf(a0.w * sc);
      af[i].us[4] = f2bf(a1.x * sc);  af[i].us[5] = f2bf(a1.y * sc);
      af[i].us[6] = f2bf(a1.z * sc);  af[i].us[7] = f2bf(a1.w * sc);
      af[i].us[8] = f2bf(a2.x * sc);  af[i].us[9] = f2bf(a2.y * sc);
      af[i].us[10] = f2bf(a2.z * sc); af[i].us[11] = f2bf(a2.w * sc);
      af[i].us[12] = f2bf(a3.x * sc); af[i].us[13] = f2bf(a3.y * sc);
      af[i].us[14] = f2bf(a3.z * sc); af[i].us[15] = f2bf(a3.w * sc);
    }
    for (int j = 0; j < 8; ++j) {
      FragS bq;
      int p = j * 16 + (lane & 15);
      const float* rp = Rp + (size_t)p * D_STATE + kk + bbase;
      for (int q4 = 0; q4 < 4; ++q4) {
        float4 v = *(const float4*)(rp + (q4 & 1) * 4 + (q4 >> 1) * 16);
        bq.us[q4 * 4 + 0] = f2bf(v.x); bq.us[q4 * 4 + 1] = f2bf(v.y);
        bq.us[q4 * 4 + 2] = f2bf(v.z); bq.us[q4 * 4 + 3] = f2bf(v.w);
      }
      for (int i = 0; i < 2; ++i) acc[i][j] = wmma_bf16(af[i].v, bq.v, acc[i][j]);
    }
  }

  // ---- Y_diag over causal s-tiles
  const float* Gp = G + (size_t)(b * NCHUNK + c) * CHUNK * CHUNK;
  const unsigned short* Xp = xdt + ((size_t)(b * SEQLEN + c * CHUNK) * NHEADS + h) * HEADDIM;
  for (int st = 0; st < CHUNK; st += 32) {
    __syncthreads();
    for (int i = 0; i < 2; ++i) {   // wide reads along p, scatter into [p][s]
      int ch = tid + i * 256;
      int ss = ch & 31, p8 = (ch >> 5) * 8;
      uint4 q = *(const uint4*)(Xp + (size_t)(st + ss) * D_INNER + p8);
      unsigned short hs[8] = {
        (unsigned short)(q.x & 0xffff), (unsigned short)(q.x >> 16),
        (unsigned short)(q.y & 0xffff), (unsigned short)(q.y >> 16),
        (unsigned short)(q.z & 0xffff), (unsigned short)(q.z >> 16),
        (unsigned short)(q.w & 0xffff), (unsigned short)(q.w >> 16)};
      for (int j = 0; j < 8; ++j) Xs[(p8 + j) * LDT + ss] = hs[j];
    }
    __syncthreads();
    if (st <= l0 + 31) {
      unsigned short* ms = &Ms[wave][0];
      float esl = __expf(-sacs[st + lane]);
      for (int r = 0; r < 32; ++r) {
        int l = l0 + r, s = st + lane;
        float g = Gp[(size_t)l * CHUNK + s];
        float m = (s <= l) ? g * __expf(sacs[l]) * esl : 0.0f;
        ms[r * LDT + lane] = f2bf(m);
      }
      FragB af2[2];
      for (int i = 0; i < 2; ++i) {
        const unsigned short* p2 = ms + (i * 16 + (lane & 15)) * LDT + abase;
        af2[i].q[0] = *(const uint4*)(p2);
        af2[i].q[1] = *(const uint4*)(p2 + 16);
      }
      for (int j = 0; j < 8; ++j) {
        FragB bfx;
        const unsigned short* xp = &Xs[(j * 16 + (lane & 15)) * LDT + bbase];
        bfx.q[0] = *(const uint4*)(xp);
        bfx.q[1] = *(const uint4*)(xp + 16);
        for (int i = 0; i < 2; ++i) acc[i][j] = wmma_bf16(af2[i].v, bfx.v, acc[i][j]);
      }
    }
  }

  float* yrow = y + (size_t)(b * SEQLEN + c * CHUNK) * D_INNER + h * HEADDIM;
  const int moff = (lane < 16) ? 0 : 8;
  for (int i = 0; i < 2; ++i)
    for (int j = 0; j < 8; ++j) {
      int lb = l0 + i * 16 + moff;
      int p = j * 16 + (lane & 15);
      for (int r = 0; r < 8; ++r)
        yrow[(size_t)(lb + r) * D_INNER + p] = acc[i][j][r];
    }
}

// silu-gate + LayerNorm; writes bf16 activations for the out-proj GEMM
__global__ __launch_bounds__(256) void k_gateln(
    const float* __restrict__ y, const float* __restrict__ zx,
    const float* __restrict__ lnw, const float* __restrict__ lnb,
    unsigned short* __restrict__ yb)
{
  int row = blockIdx.x, tid = threadIdx.x;
  __shared__ float red[256];
  const float* zr = zx + (size_t)row * D_IN_PROJ;
  const float* yr = y + (size_t)row * D_INNER;
  unsigned short* yo = yb + (size_t)row * D_INNER;
  float v[16], s = 0.0f;
  for (int i = 0; i < 16; ++i) {
    int idx = tid + i * 256;
    float t = yr[idx] * siluf_(zr[idx]);
    v[i] = t; s += t;
  }
  red[tid] = s; __syncthreads();
  for (int o = 128; o > 0; o >>= 1) { if (tid < o) red[tid] += red[tid + o]; __syncthreads(); }
  float mu = red[0] / (float)D_INNER;
  __syncthreads();
  float q = 0.0f;
  for (int i = 0; i < 16; ++i) { float d = v[i] - mu; q += d * d; }
  red[tid] = q; __syncthreads();
  for (int o = 128; o > 0; o >>= 1) { if (tid < o) red[tid] += red[tid + o]; __syncthreads(); }
  float inv = rsqrtf(red[0] / (float)D_INNER + LN_EPS);
  for (int i = 0; i < 16; ++i) {
    int idx = tid + i * 256;
    yo[idx] = f2bf((v[i] - mu) * inv * lnw[idx] + lnb[idx]);
  }
}

// ---------------------------------------------------------------------------
extern "C" void kernel_launch(void* const* d_in, const int* in_sizes, int n_in,
                              void* d_out, int out_size, void* d_ws, size_t ws_size,
                              hipStream_t stream) {
  const float* u       = (const float*)d_in[0];
  const float* in_w    = (const float*)d_in[1];
  const float* in_b    = (const float*)d_in[2];
  const float* conv_w  = (const float*)d_in[3];
  const float* conv_b  = (const float*)d_in[4];
  const float* dt_bias = (const float*)d_in[5];
  const float* A_log   = (const float*)d_in[6];
  const float* ln_w    = (const float*)d_in[7];
  const float* ln_b    = (const float*)d_in[8];
  const float* out_w   = (const float*)d_in[9];
  const float* out_b   = (const float*)d_in[10];
  float* out = (float*)d_out;

  char* wp = (char*)d_ws;
  auto alloc = [&](size_t bytes) {
    char* p = wp; wp += (bytes + 255) & ~(size_t)255; return p;
  };
  float* zx   = (float*)alloc((size_t)ROWS * D_IN_PROJ * 4);
  float* xbc  = (float*)alloc((size_t)ROWS * CONV_DIM * 4);
  float* dt   = (float*)alloc((size_t)ROWS * NHEADS * 4);
  float* acs  = (float*)alloc((size_t)BATCH * NHEADS * NCHUNK * CHUNK * 4);
  float* alast= (float*)alloc((size_t)BATCH * NHEADS * NCHUNK * 4);
  unsigned short* xdt = (unsigned short*)alloc((size_t)ROWS * D_INNER * 2);
  float* G    = (float*)alloc((size_t)BATCH * NCHUNK * CHUNK * CHUNK * 4);
  float* cs   = (float*)alloc((size_t)BATCH * NCHUNK * NHEADS * HEADDIM * D_STATE * 4);
  float* R    = (float*)alloc((size_t)BATCH * NCHUNK * NHEADS * HEADDIM * D_STATE * 4);
  float* y    = (float*)alloc((size_t)ROWS * D_INNER * 4);
  unsigned short* ub  = (unsigned short*)alloc((size_t)ROWS * D_MODEL * 2);
  unsigned short* wib = (unsigned short*)alloc((size_t)D_IN_PROJ * D_MODEL * 2);
  unsigned short* wob = (unsigned short*)alloc((size_t)D_MODEL * D_INNER * 2);
  unsigned short* yb  = (unsigned short*)alloc((size_t)ROWS * D_INNER * 2);

  // bf16 staging of GEMM operands
  size_t n4u = (size_t)ROWS * D_MODEL / 4;
  k_cvt<<<(unsigned)((n4u + 255) / 256), 256, 0, stream>>>(u, ub, n4u);
  size_t n4wi = (size_t)D_IN_PROJ * D_MODEL / 4;
  k_cvt<<<(unsigned)((n4wi + 255) / 256), 256, 0, stream>>>(in_w, wib, n4wi);
  size_t n4wo = (size_t)D_MODEL * D_INNER / 4;
  k_cvt<<<(unsigned)((n4wo + 255) / 256), 256, 0, stream>>>(out_w, wob, n4wo);

  dim3 g1((D_IN_PROJ + 127) / 128, ROWS / 128);
  k_gemm_bf<<<g1, 256, 0, stream>>>(ub, D_MODEL, wib, D_MODEL, in_b,
                                    zx, D_IN_PROJ, ROWS, D_IN_PROJ, D_MODEL);
  k_conv<<<(unsigned)(((size_t)ROWS * CONV_DIM + 255) / 256), 256, 0, stream>>>(
      zx, conv_w, conv_b, xbc);
  k_dt<<<(ROWS * NHEADS + 255) / 256, 256, 0, stream>>>(zx, dt_bias, dt);
  k_acs<<<BATCH * NHEADS * NCHUNK, CHUNK, 0, stream>>>(dt, A_log, acs, alast);
  k_xdt<<<(unsigned)(((size_t)ROWS * D_INNER + 255) / 256), 256, 0, stream>>>(
      xbc, dt, xdt);
  k_G<<<dim3(BATCH * NCHUNK, 4), 256, 0, stream>>>(xbc, G);
  k_cs<<<BATCH * NCHUNK * NHEADS, 256, 0, stream>>>(xdt, xbc, acs, alast, cs);
  k_recur<<<(BATCH * NHEADS * HEADDIM * D_STATE + 255) / 256, 256, 0, stream>>>(
      cs, alast, R);
  k_y<<<BATCH * NCHUNK * NHEADS, 256, 0, stream>>>(xdt, xbc, G, R, acs, y);
  k_gateln<<<ROWS, 256, 0, stream>>>(y, zx, ln_w, ln_b, yb);
  dim3 g2(D_MODEL / 128, ROWS / 128);
  k_gemm_bf<<<g2, 256, 0, stream>>>(yb, D_INNER, wob, D_INNER, out_b,
                                    out, D_MODEL, ROWS, D_MODEL, D_INNER);
}
